// RankMixerLayer_72533407695295
// MI455X (gfx1250) — matrix operand order repaired and checked by reading, not compiled, not run on hardware
//
#include <hip/hip_runtime.h>
#include <stdint.h>

#define B_   2048
#define T_   16
#define D_   1024
#define DF_  4096
#define EPS_ 1e-6f

// Flip to 0 if the async-to-LDS mnemonic is ever rejected by the assembler.
#ifndef USE_ASYNC_COPY
#define USE_ASYNC_COPY 1
#endif

typedef __attribute__((ext_vector_type(16))) __bf16          v16bf;
typedef __attribute__((ext_vector_type(8)))  __bf16          v8bf;
typedef __attribute__((ext_vector_type(8)))  float           v8f;
typedef __attribute__((ext_vector_type(8)))  unsigned short  v8us;

__device__ __forceinline__ float bf16u_to_f32(unsigned short h) {
  union { uint32_t u; float f; } c; c.u = ((uint32_t)h) << 16;
  return c.f;
}
__device__ __forceinline__ unsigned short bf_as_us(__bf16 b) {
  return __builtin_bit_cast(unsigned short, b);
}

__device__ __forceinline__ float wave_sum(float v) {
#pragma unroll
  for (int m = 16; m >= 1; m >>= 1) v += __shfl_xor(v, m, 32);
  return v;
}

// A fragment (16x32 bf16): lane holds M = lane&15; K runs [hx*8, hx*8+8) and
// [16+hx*8, 16+hx*8+8) -> two contiguous 16B loads from a bf16 row.
__device__ __forceinline__ v16bf load_a_frag(const unsigned short* rowk0, int hx) {
  const v8us lo = *(const v8us*)(rowk0 + hx * 8);
  const v8us hi = *(const v8us*)(rowk0 + 16 + hx * 8);
  const v8bf l = __builtin_bit_cast(v8bf, lo);
  const v8bf h = __builtin_bit_cast(v8bf, hi);
  return __builtin_shufflevector(l, h, 0,1,2,3,4,5,6,7,8,9,10,11,12,13,14,15);
}

// B fragment (32x16): lane N = col; K = hx*8 + 2v (+1) (+8 for v>=4); fp32 tile in LDS,
// row-major [k][NC], paired f32->bf16 conversion (v_cvt_pk_bf16_f32).
template <int NC>
__device__ __forceinline__ v16bf load_b_frag(const float* Wt, int col, int hx) {
  v16bf b;
#pragma unroll
  for (int v = 0; v < 8; ++v) {
    const int k = (hx << 3) + (v << 1) + ((v >> 2) << 3);
    b[2 * v]     = (__bf16)Wt[k * NC + col];
    b[2 * v + 1] = (__bf16)Wt[(k + 1) * NC + col];
  }
  return b;
}

#if USE_ASYNC_COPY
#define STAGE_B128(dst, src)                                                    \
  asm volatile("global_load_async_to_lds_b128 %0, %1, off"                      \
               :: "v"((uint32_t)(uintptr_t)(dst)), "v"(src) : "memory")
#define STAGE_WAIT() asm volatile("s_wait_asynccnt 0x0" ::: "memory")
#else
#define STAGE_B128(dst, src) *(float4*)(dst) = *(const float4*)(src)
#define STAGE_WAIT() ((void)0)
#endif

// 4-WMMA burst with constant reuse-B hints (reuse_b must be a literal).
#define WMMA4()                                                                          \
  do {                                                                                   \
    acc[0] = __builtin_amdgcn_wmma_f32_16x16x32_bf16(false, afr[0], false, bfrag,        \
                                                     (short)0, acc[0], false, true);     \
    acc[1] = __builtin_amdgcn_wmma_f32_16x16x32_bf16(false, afr[1], false, bfrag,        \
                                                     (short)0, acc[1], false, true);     \
    acc[2] = __builtin_amdgcn_wmma_f32_16x16x32_bf16(false, afr[2], false, bfrag,        \
                                                     (short)0, acc[2], false, true);     \
    acc[3] = __builtin_amdgcn_wmma_f32_16x16x32_bf16(false, afr[3], false, bfrag,        \
                                                     (short)0, acc[3], false, false);    \
  } while (0)

// ---------------------------------------------------------------------------
// Kernel 1: x1 = LayerNorm(x + token_mix(x)) * g1 + be1 -> bf16 in ws.
// token_mix: mixed[b,t,d] = x[b, d>>6, t*64 + (d&63)].  Block per (b,t).
// ---------------------------------------------------------------------------
__global__ __launch_bounds__(256) void k_ln1(const float* __restrict__ x,
                                             const float* __restrict__ g1,
                                             const float* __restrict__ be1,
                                             unsigned short* __restrict__ x1bf) {
  const int idx = blockIdx.x;
  const int b   = idx >> 4;
  const int t   = idx & 15;
  const int tid = threadIdx.x;
  const int d0  = tid * 4;

  const float4 xs = *(const float4*)(x + (size_t)idx * D_ + d0);
  const float4 ms = *(const float4*)(x + ((size_t)b * T_ + (d0 >> 6)) * D_ + t * 64 + (d0 & 63));
  float s[4] = { xs.x + ms.x, xs.y + ms.y, xs.z + ms.z, xs.w + ms.w };

  float ls = s[0] + s[1] + s[2] + s[3];
  float lq = s[0]*s[0] + s[1]*s[1] + s[2]*s[2] + s[3]*s[3];
  ls = wave_sum(ls);
  lq = wave_sum(lq);

  __shared__ float red[16];
  __shared__ float stat[2];
  const int wv = tid >> 5, ln = tid & 31;
  if (ln == 0) { red[wv * 2] = ls; red[wv * 2 + 1] = lq; }
  __syncthreads();
  if (tid == 0) {
    float ts = 0.f, tq = 0.f;
#pragma unroll
    for (int i = 0; i < 8; ++i) { ts += red[i * 2]; tq += red[i * 2 + 1]; }
    const float mean = ts * (1.0f / D_);
    const float var  = tq * (1.0f / D_) - mean * mean;
    stat[0] = mean;
    stat[1] = rsqrtf(var + EPS_);
  }
  __syncthreads();
  const float mean = stat[0], rstd = stat[1];

  ushort4 o;
  o.x = bf_as_us((__bf16)((s[0] - mean) * rstd * g1[d0 + 0] + be1[d0 + 0]));
  o.y = bf_as_us((__bf16)((s[1] - mean) * rstd * g1[d0 + 1] + be1[d0 + 1]));
  o.z = bf_as_us((__bf16)((s[2] - mean) * rstd * g1[d0 + 2] + be1[d0 + 2]));
  o.w = bf_as_us((__bf16)((s[3] - mean) * rstd * g1[d0 + 3] + be1[d0 + 3]));
  *(ushort4*)(x1bf + (size_t)idx * D_ + d0) = o;
}

// ---------------------------------------------------------------------------
// Kernel 2: H[t,b,f] = gelu( x1[b,t,:] @ W1[t,:,f] + b1[t,f] ) -> bf16.
// 128 threads (4 waves); tile 64 rows x 64 cols; K = D_ step 32.
// Double-buffered async weight staging; A direct from global (b128 pairs).
// ---------------------------------------------------------------------------
__global__ __launch_bounds__(128) void k_ffn1(const unsigned short* __restrict__ x1bf,
                                              const float* __restrict__ W1,
                                              const float* __restrict__ b1,
                                              unsigned short* __restrict__ Hbuf) {
  const int t   = blockIdx.z;
  const int r0  = blockIdx.y * 64;
  const int fb0 = blockIdx.x * 64;
  const int tid = threadIdx.x;
  const int wv  = tid >> 5;
  const int ln  = tid & 31;
  const int hx  = ln >> 4;
  const int n   = ln & 15;
  const int col  = wv * 16 + n;
  const int gcol = fb0 + col;

  __shared__ float Wt[2][32 * 64];   // 2 x 8 KB, double buffered

  v8f acc[4];
#pragma unroll
  for (int i = 0; i < 4; ++i) acc[i] = (v8f){0.f,0.f,0.f,0.f,0.f,0.f,0.f,0.f};

  const unsigned short* aRow[4];
#pragma unroll
  for (int i = 0; i < 4; ++i)
    aRow[i] = x1bf + ((size_t)(r0 + i * 16 + (ln & 15)) * T_ + t) * D_;

  // staging: 32x64 fp32 chunk = 512 b128; thread owns (kbase + 8j, cc)
  const int kbase = tid >> 4;
  const int cc    = (tid & 15) << 2;
  const float* srcp = &W1[((size_t)t * D_ + (size_t)kbase) * DF_ + fb0 + cc];
  const int dstoff = kbase * 64 + cc;

  // prologue: stage chunk 0 into buffer 0
#pragma unroll
  for (int j = 0; j < 4; ++j) STAGE_B128(&Wt[0][dstoff + j * (8 * 64)], srcp + (size_t)j * 8 * DF_);
  srcp += (size_t)32 * DF_;
  STAGE_WAIT();
  __syncthreads();

  int sel = 0;
  for (int k0 = 0; k0 < D_; k0 += 32) {
    if (k0 + 32 < D_) {
#pragma unroll
      for (int j = 0; j < 4; ++j)
        STAGE_B128(&Wt[sel ^ 1][dstoff + j * (8 * 64)], srcp + (size_t)j * 8 * DF_);
      srcp += (size_t)32 * DF_;
    }

    const v16bf bfrag = load_b_frag<64>(Wt[sel], col, hx);
    v16bf afr[4];
#pragma unroll
    for (int i = 0; i < 4; ++i) afr[i] = load_a_frag(aRow[i] + k0, hx);
    WMMA4();

    STAGE_WAIT();
    __syncthreads();
    sel ^= 1;
  }

  const float bias = b1[t * DF_ + gcol];
#pragma unroll
  for (int i = 0; i < 4; ++i) {
#pragma unroll
    for (int r = 0; r < 8; ++r) {
      const int brow = r0 + i * 16 + (hx << 3) + r;
      const float hv = acc[i][r] + bias;
      const float gv = 0.5f * hv * (1.0f + erff(hv * 0.70710678118654752f));
      Hbuf[((size_t)t * B_ + brow) * DF_ + gcol] = bf_as_us((__bf16)gv);
    }
  }
}

// ---------------------------------------------------------------------------
// Kernel 3: y = H[t,b,:] @ W2[t,:,d] + b2 + x1  -> fp32 out (pre-LN2).
// 256 threads (8 waves); tile 64 rows x 128 cols; K = DF_ step 32.
// ---------------------------------------------------------------------------
__global__ __launch_bounds__(256) void k_ffn2(const unsigned short* __restrict__ Hbuf,
                                              const float* __restrict__ W2,
                                              const float* __restrict__ b2,
                                              const unsigned short* __restrict__ x1bf,
                                              float* __restrict__ out) {
  const int t   = blockIdx.z;
  const int r0  = blockIdx.y * 64;
  const int nb0 = blockIdx.x * 128;
  const int tid = threadIdx.x;
  const int wv  = tid >> 5;
  const int ln  = tid & 31;
  const int hx  = ln >> 4;
  const int n   = ln & 15;
  const int col  = wv * 16 + n;
  const int gcol = nb0 + col;

  __shared__ float Wt[2][32 * 128];  // 2 x 16 KB, double buffered

  v8f acc[4];
#pragma unroll
  for (int i = 0; i < 4; ++i) acc[i] = (v8f){0.f,0.f,0.f,0.f,0.f,0.f,0.f,0.f};

  const unsigned short* aRow[4];
#pragma unroll
  for (int i = 0; i < 4; ++i)
    aRow[i] = Hbuf + ((size_t)t * B_ + (size_t)(r0 + i * 16 + (ln & 15))) * DF_;

  // staging: 32x128 fp32 chunk = 1024 b128; thread owns (kbase + 8j, cc)
  const int kbase = tid >> 5;
  const int cc    = (tid & 31) << 2;
  const float* srcp = &W2[((size_t)t * DF_ + (size_t)kbase) * D_ + nb0 + cc];
  const int dstoff = kbase * 128 + cc;

#pragma unroll
  for (int j = 0; j < 4; ++j) STAGE_B128(&Wt[0][dstoff + j * (8 * 128)], srcp + (size_t)j * 8 * D_);
  srcp += (size_t)32 * D_;
  STAGE_WAIT();
  __syncthreads();

  int sel = 0;
  for (int k0 = 0; k0 < DF_; k0 += 32) {
    if (k0 + 32 < DF_) {
#pragma unroll
      for (int j = 0; j < 4; ++j)
        STAGE_B128(&Wt[sel ^ 1][dstoff + j * (8 * 128)], srcp + (size_t)j * 8 * D_);
      srcp += (size_t)32 * D_;
    }

    const v16bf bfrag = load_b_frag<128>(Wt[sel], col, hx);
    v16bf afr[4];
#pragma unroll
    for (int i = 0; i < 4; ++i) afr[i] = load_a_frag(aRow[i] + k0, hx);
    WMMA4();

    STAGE_WAIT();
    __syncthreads();
    sel ^= 1;
  }

  const float bias = b2[t * D_ + gcol];
#pragma unroll
  for (int i = 0; i < 4; ++i) {
#pragma unroll
    for (int r = 0; r < 8; ++r) {
      const int brow = r0 + i * 16 + (hx << 3) + r;
      const size_t o = ((size_t)brow * T_ + t) * D_ + gcol;
      out[o] = acc[i][r] + bias + bf16u_to_f32(x1bf[o]);
    }
  }
}

// ---------------------------------------------------------------------------
// Kernel 4: in-place row LayerNorm on d_out with g2/be2.
// ---------------------------------------------------------------------------
__global__ __launch_bounds__(256) void k_ln2(float* __restrict__ y,
                                             const float* __restrict__ g2,
                                             const float* __restrict__ be2) {
  const int idx = blockIdx.x;
  const int tid = threadIdx.x;
  const int d0  = tid * 4;

  const float4 ys = *(const float4*)(y + (size_t)idx * D_ + d0);
  float s[4] = { ys.x, ys.y, ys.z, ys.w };

  float ls = s[0] + s[1] + s[2] + s[3];
  float lq = s[0]*s[0] + s[1]*s[1] + s[2]*s[2] + s[3]*s[3];
  ls = wave_sum(ls);
  lq = wave_sum(lq);

  __shared__ float red[16];
  __shared__ float stat[2];
  const int wv = tid >> 5, ln = tid & 31;
  if (ln == 0) { red[wv * 2] = ls; red[wv * 2 + 1] = lq; }
  __syncthreads();
  if (tid == 0) {
    float ts = 0.f, tq = 0.f;
#pragma unroll
    for (int i = 0; i < 8; ++i) { ts += red[i * 2]; tq += red[i * 2 + 1]; }
    const float mean = ts * (1.0f / D_);
    const float var  = tq * (1.0f / D_) - mean * mean;
    stat[0] = mean;
    stat[1] = rsqrtf(var + EPS_);
  }
  __syncthreads();
  const float mean = stat[0], rstd = stat[1];

  float4 o;
  o.x = (s[0] - mean) * rstd * g2[d0 + 0] + be2[d0 + 0];
  o.y = (s[1] - mean) * rstd * g2[d0 + 1] + be2[d0 + 1];
  o.z = (s[2] - mean) * rstd * g2[d0 + 2] + be2[d0 + 2];
  o.w = (s[3] - mean) * rstd * g2[d0 + 3] + be2[d0 + 3];
  *(float4*)(y + (size_t)idx * D_ + d0) = o;
}

// ---------------------------------------------------------------------------
extern "C" void kernel_launch(void* const* d_in, const int* in_sizes, int n_in,
                              void* d_out, int out_size, void* d_ws, size_t ws_size,
                              hipStream_t stream) {
  const float* x   = (const float*)d_in[0];
  const float* W1  = (const float*)d_in[1];
  const float* b1  = (const float*)d_in[2];
  const float* W2  = (const float*)d_in[3];
  const float* b2  = (const float*)d_in[4];
  const float* g1  = (const float*)d_in[5];
  const float* be1 = (const float*)d_in[6];
  const float* g2  = (const float*)d_in[7];
  const float* be2 = (const float*)d_in[8];
  float* out = (float*)d_out;

  unsigned short* x1bf = (unsigned short*)d_ws;         // B*T*D bf16  (64 MB)
  unsigned short* Hbuf = x1bf + (size_t)B_ * T_ * D_;   // T*B*DF bf16 (256 MB)

  k_ln1<<<B_ * T_, 256, 0, stream>>>(x, g1, be1, x1bf);
  k_ffn1<<<dim3(DF_ / 64, B_ / 64, T_), 128, 0, stream>>>(x1bf, W1, b1, Hbuf);
  k_ffn2<<<dim3(D_ / 128, B_ / 64, T_), 256, 0, stream>>>(Hbuf, W2, b2, x1bf, out);
  k_ln2<<<B_ * T_, 256, 0, stream>>>(out, g2, be2);
}
